// WeightPopupLayer_56315611185241
// MI455X (gfx1250) — matrix operand order repaired and checked by reading, not compiled, not run on hardware
//
#include <hip/hip_runtime.h>

// ---------------------------------------------------------------------------
// WeightPopupLayer on MI455X (gfx1250, wave32):
//   1) global radix-select threshold over |scores| (keep top 50%)
//   2) mask weights -> bf16 hi/lo (ws), pre-split x -> bf16 hi/lo (ws)
//   3) bf16 WMMA GEMM: out = x @ (W*mask)^T + bias, fp32 accuracy via
//      hi*hi + hi*lo + lo*hi. B tiles staged to LDS with
//      global_load_async_to_lds_b128 (double-buffered, ASYNCcnt).
// ---------------------------------------------------------------------------

#define BATCH   8192
#define NOUT    4096
#define KIN     4096
#define NSCORE  (4096u * 4096u)
#define NX      (8192u * 4096u)
#define RANK_J  8388608u   // int((1-K)*n), K=0.5

typedef __attribute__((ext_vector_type(16))) __bf16          v16bf;
typedef __attribute__((ext_vector_type(16))) unsigned short  v16u;
typedef __attribute__((ext_vector_type(8)))  unsigned short  v8u;
typedef __attribute__((ext_vector_type(8)))  float           v8f;

struct SelState {
    unsigned prefix;
    unsigned rank;
    unsigned hist[256];
    float    thresh;
};

__device__ inline unsigned short f2bf_rne(float f) {
    unsigned u = __float_as_uint(f);
    unsigned r = 0x7FFFu + ((u >> 16) & 1u);
    return (unsigned short)((u + r) >> 16);
}
__device__ inline float bf2f(unsigned short b) {
    return __uint_as_float(((unsigned)b) << 16);
}

// -------------------------- radix select (threshold) -----------------------

__global__ void rs_init(SelState* st) {
    if (threadIdx.x < 256) st->hist[threadIdx.x] = 0u;
    if (threadIdx.x == 0) { st->prefix = 0u; st->rank = RANK_J; st->thresh = 0.f; }
}

__global__ void rs_hist(const float* __restrict__ scores, SelState* __restrict__ st, int shift) {
    __shared__ unsigned lh[256];
    if (threadIdx.x < 256) lh[threadIdx.x] = 0u;
    __syncthreads();
    const unsigned prefix = st->prefix;
    const int hs = shift + 8;
    unsigned idx    = blockIdx.x * blockDim.x + threadIdx.x;
    unsigned stride = gridDim.x * blockDim.x;
    for (; idx < NSCORE; idx += stride) {
        unsigned bits = __float_as_uint(fabsf(scores[idx]));
        bool ok = (hs >= 32) ? true : ((bits >> hs) == (prefix >> hs));
        if (ok) atomicAdd(&lh[(bits >> shift) & 0xFFu], 1u);
    }
    __syncthreads();
    if (threadIdx.x < 256) {
        unsigned c = lh[threadIdx.x];
        if (c) atomicAdd(&st->hist[threadIdx.x], c);
    }
}

__global__ void rs_select(SelState* st, int shift, int final_pass) {
    if (threadIdx.x != 0 || blockIdx.x != 0) return;
    unsigned rank = st->rank, cum = 0u;
    int d = 0;
    for (; d < 256; ++d) {
        unsigned c = st->hist[d];
        if (cum + c > rank) break;
        cum += c;
    }
    if (d > 255) d = 255;
    st->rank = rank - cum;
    st->prefix |= ((unsigned)d) << shift;
    for (int i = 0; i < 256; ++i) st->hist[i] = 0u;
    if (final_pass) st->thresh = __uint_as_float(st->prefix);
}

// ------------------- precision-split conversion passes ---------------------

__global__ void mask_convert_w(const float* __restrict__ w,
                               const float* __restrict__ scores,
                               const SelState* __restrict__ st,
                               unsigned short* __restrict__ whi,
                               unsigned short* __restrict__ wlo) {
    const unsigned tbits = st->prefix;   // keep iff bits(|score|) >= threshold
    unsigned idx    = blockIdx.x * blockDim.x + threadIdx.x;
    unsigned stride = gridDim.x * blockDim.x;
    for (; idx < NSCORE; idx += stride) {
        unsigned sb = __float_as_uint(fabsf(scores[idx]));
        float v = (sb >= tbits) ? w[idx] : 0.0f;
        unsigned short h = f2bf_rne(v);
        whi[idx] = h;
        wlo[idx] = f2bf_rne(v - bf2f(h));
    }
}

__global__ void convert_x(const float* __restrict__ x,
                          unsigned short* __restrict__ xhi,
                          unsigned short* __restrict__ xlo) {
    unsigned idx    = blockIdx.x * blockDim.x + threadIdx.x;
    unsigned stride = gridDim.x * blockDim.x;
    for (; idx < NX; idx += stride) {
        float v = x[idx];
        unsigned short h = f2bf_rne(v);
        xhi[idx] = h;
        xlo[idx] = f2bf_rne(v - bf2f(h));
    }
}

// ------------------------------- WMMA GEMM ---------------------------------

__device__ inline v8f wmma_bf16(v16u a, v16u b, v8f c) {
    union { v16u u; v16bf b; } ua, ub;
    ua.u = a; ub.u = b;
    return __builtin_amdgcn_wmma_f32_16x16x32_bf16(
        false, ua.b, false, ub.b, (short)0, c, false, false);
}

// A fragment (16x32 bf16, row-major source): lane = (half, l16) covers row
// l16, K runs [16*half, +16).
__device__ inline v16u ldA(const unsigned short* p) {
    v8u a = *(const v8u*)p;
    v8u b = *(const v8u*)(p + 8);
    v16u r;
    #pragma unroll
    for (int i = 0; i < 8; ++i) { r[i] = a[i]; r[i + 8] = b[i]; }
    return r;
}

__device__ inline void async_b128(unsigned lds_off, const void* gaddr) {
    asm volatile("global_load_async_to_lds_b128 %0, %1, off"
                 :: "v"(lds_off), "v"(gaddr) : "memory");
}
__device__ inline void wait_async0() {
    asm volatile("s_wait_asynccnt 0" ::: "memory");
}

// Block: 256 rows x 64 cols of C. 8 waves, each wave a 32x64 tile.
// B (hi/lo) 64x32-K tiles double-buffered in LDS via async copies.
__global__ __launch_bounds__(256)
void gemm_wmma(const unsigned short* __restrict__ xhi,
               const unsigned short* __restrict__ xlo,
               const unsigned short* __restrict__ whi,
               const unsigned short* __restrict__ wlo,
               const float* __restrict__ bias,
               float* __restrict__ out) {
    __shared__ unsigned short lds_bh[2][64 * 32];   // 8 KB
    __shared__ unsigned short lds_bl[2][64 * 32];   // 8 KB

    const int tid  = threadIdx.x;
    const int wave = tid >> 5;
    const int lane = tid & 31;
    const int half = lane >> 4;
    const int l16  = lane & 15;

    const int nStrip = blockIdx.x & 63;     // NOUT/64 strips
    const int mBlk   = blockIdx.x >> 6;     // BATCH/256 row blocks
    const int n0 = nStrip * 64;
    const int m0 = mBlk * 256 + wave * 32;

    // staging: thread copies 16B of column (n0 + tid/4), k offset (tid%4)*8
    const int sc = tid >> 2;
    const int sk = (tid & 3) * 8;
    const unsigned short* ghi = whi + (size_t)(n0 + sc) * KIN + sk;
    const unsigned short* glo = wlo + (size_t)(n0 + sc) * KIN + sk;

    // A row pointers: two 16-row M subtiles
    const unsigned short* ah0 = xhi + (size_t)(m0 + l16) * KIN + half * 16;
    const unsigned short* al0 = xlo + (size_t)(m0 + l16) * KIN + half * 16;
    const unsigned short* ah1 = ah0 + (size_t)16 * KIN;
    const unsigned short* al1 = al0 + (size_t)16 * KIN;

    v8f acc[2][4];
    #pragma unroll
    for (int m = 0; m < 2; ++m)
        #pragma unroll
        for (int t = 0; t < 4; ++t) acc[m][t] = (v8f){};

    // prologue: stage K tile 0 into buffer 0
    async_b128((unsigned)(unsigned long long)&lds_bh[0][tid * 8], (const void*)ghi);
    async_b128((unsigned)(unsigned long long)&lds_bl[0][tid * 8], (const void*)glo);
    wait_async0();
    __syncthreads();

    for (int k0 = 0; k0 < KIN; k0 += 32) {
        const int buf = (k0 >> 5) & 1;

        if (k0 + 32 < KIN) {   // stage next tile into other buffer (overlapped)
            async_b128((unsigned)(unsigned long long)&lds_bh[buf ^ 1][tid * 8],
                       (const void*)(ghi + k0 + 32));
            async_b128((unsigned)(unsigned long long)&lds_bl[buf ^ 1][tid * 8],
                       (const void*)(glo + k0 + 32));
            __builtin_prefetch((const void*)(ah0 + k0 + 32), 0, 0);
        }

        // A fragments (pre-split bf16, direct global loads)
        v16u a0h = ldA(ah0 + k0);
        v16u a0l = ldA(al0 + k0);
        v16u a1h = ldA(ah1 + k0);
        v16u a1l = ldA(al1 + k0);

        #pragma unroll
        for (int t = 0; t < 4; ++t) {
            // B fragment from LDS: col 16t+l16, contiguous K run at 16*half
            const unsigned short* pbh = &lds_bh[buf][(16 * t + l16) * 32 + half * 16];
            const unsigned short* pbl = &lds_bl[buf][(16 * t + l16) * 32 + half * 16];
            v16u bh = *(const v16u*)pbh;
            v16u bl = *(const v16u*)pbl;
            acc[0][t] = wmma_bf16(a0h, bh, acc[0][t]);
            acc[0][t] = wmma_bf16(a0h, bl, acc[0][t]);
            acc[0][t] = wmma_bf16(a0l, bh, acc[0][t]);
            acc[1][t] = wmma_bf16(a1h, bh, acc[1][t]);
            acc[1][t] = wmma_bf16(a1h, bl, acc[1][t]);
            acc[1][t] = wmma_bf16(a1l, bh, acc[1][t]);
        }

        wait_async0();      // own copies done before barrier
        __syncthreads();    // all waves: reads done, next buffer filled
    }

    // epilogue: C layout -> row = m0 + 16*ms + 8*half + r, col = n0 + 16t + l16
    #pragma unroll
    for (int ms = 0; ms < 2; ++ms) {
        #pragma unroll
        for (int t = 0; t < 4; ++t) {
            const int col = n0 + 16 * t + l16;
            const float bv = bias[col];
            #pragma unroll
            for (int r = 0; r < 8; ++r) {
                const int row = m0 + 16 * ms + half * 8 + r;
                out[(size_t)row * NOUT + col] = acc[ms][t][r] + bv;
            }
        }
    }
}

// ------------------------------- launcher ----------------------------------

extern "C" void kernel_launch(void* const* d_in, const int* in_sizes, int n_in,
                              void* d_out, int out_size, void* d_ws, size_t ws_size,
                              hipStream_t stream) {
    (void)in_sizes; (void)n_in; (void)out_size; (void)ws_size;

    const float* x      = (const float*)d_in[0];
    const float* w      = (const float*)d_in[1];
    const float* bias   = (const float*)d_in[2];
    const float* scores = (const float*)d_in[3];
    float* out = (float*)d_out;

    SelState* st = (SelState*)d_ws;
    unsigned short* whi = (unsigned short*)((char*)d_ws + 1024);
    unsigned short* wlo = whi + NSCORE;   // 32 MB each
    unsigned short* xhi = wlo + NSCORE;   // 64 MB each
    unsigned short* xlo = xhi + NX;

    // 1) global threshold via 4-pass radix select on |score| bits
    hipLaunchKernelGGL(rs_init, dim3(1), dim3(256), 0, stream, st);
    const int shifts[4] = {24, 16, 8, 0};
    for (int p = 0; p < 4; ++p) {
        hipLaunchKernelGGL(rs_hist, dim3(2048), dim3(256), 0, stream, scores, st, shifts[p]);
        hipLaunchKernelGGL(rs_select, dim3(1), dim3(1), 0, stream, st, shifts[p], (int)(shifts[p] == 0));
    }

    // 2) mask + bf16 hi/lo conversion of weights and x
    hipLaunchKernelGGL(mask_convert_w, dim3(4096), dim3(256), 0, stream,
                       w, scores, st, whi, wlo);
    hipLaunchKernelGGL(convert_x, dim3(8192), dim3(256), 0, stream, x, xhi, xlo);

    // 3) WMMA GEMM: 32 row-blocks * 64 col-strips = 2048 blocks, 8 waves each
    hipLaunchKernelGGL(gemm_wmma, dim3(2048), dim3(256), 0, stream,
                       xhi, xlo, whi, wlo, bias, out);
}
